// auto_attn_block_56229711839935
// MI455X (gfx1250) — matrix operand order repaired and checked by reading, not compiled, not run on hardware
//
#include <hip/hip_runtime.h>
#include <hip/hip_bf16.h>
#include <math.h>

// ---- problem constants ----
#define B_  32
#define D_  512
#define C_  512
#define P_  64
#define N_  (B_ * C_)      // 16384 rows (b*C + c)
#define NS_ 8              // number of shifts = D_/P_
#define KT_ 64             // GEMM K-tile

typedef __bf16 bf16_t;
typedef __attribute__((ext_vector_type(16))) __bf16 v16bf;
typedef __attribute__((ext_vector_type(8)))  float  v8f;

union Frag16 { uint4 u[2]; v16bf v; };

// CDNA5 async global->LDS copy path (guarded; falls back to load+ds_store)
#if defined(__has_builtin)
# if __has_builtin(__builtin_amdgcn_global_load_async_to_lds_b128) && \
     __has_builtin(__builtin_amdgcn_s_wait_asynccnt)
#  define USE_ASYNC_LDS 1
# endif
#endif
#ifndef USE_ASYNC_LDS
# define USE_ASYNC_LDS 0
#endif

#if USE_ASYNC_LDS
// builtin signature: (int4 addrspace(1)*, int4 addrspace(3)*, imm offset, imm cpol)
typedef int v4i_t __attribute__((ext_vector_type(4)));
typedef __attribute__((address_space(1))) v4i_t* gv4i_p;
typedef __attribute__((address_space(3))) v4i_t* lv4i_p;
#endif

// ============================================================
// Weight preparation kernels (tiny; GEMM-folded weights in bf16)
// ============================================================
// Wqk[d][k] = sum_e wq[e][k] * wk[e][d]   (so QK = X @ Wqk^T)
__global__ __launch_bounds__(256) void k_prep_wqk(const float* __restrict__ wq,
                                                  const float* __restrict__ wk,
                                                  bf16_t* __restrict__ out) {
    int idx = blockIdx.x * 256 + threadIdx.x;  // D_*D_
    int d = idx >> 9, k = idx & 511;
    float s = 0.f;
    for (int e = 0; e < D_; ++e) s += wq[e * D_ + k] * wk[e * D_ + d];
    out[idx] = (bf16_t)s;
}

// Wvo[e'][d] = sum_e wo[e'][e] * wv[e][d]  (so H = U @ Wvo^T)
__global__ __launch_bounds__(256) void k_prep_wvo(const float* __restrict__ wo,
                                                  const float* __restrict__ wv,
                                                  bf16_t* __restrict__ out) {
    int idx = blockIdx.x * 256 + threadIdx.x;
    int ep = idx >> 9, d = idx & 511;
    float s = 0.f;
    for (int e = 0; e < D_; ++e) s += wo[ep * D_ + e] * wv[e * D_ + d];
    out[idx] = (bf16_t)s;
}

// bqk[d] = sum_e bq[e]*wk[e][d] ;  bvo[e'] = bo[e'] + sum_e wo[e'][e]*bv[e]
__global__ __launch_bounds__(256) void k_prep_bias(const float* __restrict__ bq,
                                                   const float* __restrict__ wk,
                                                   const float* __restrict__ wo,
                                                   const float* __restrict__ bv,
                                                   const float* __restrict__ bo,
                                                   float* __restrict__ bqk,
                                                   float* __restrict__ bvo) {
    int i = blockIdx.x * 256 + threadIdx.x;   // 2*D_
    if (i < D_) {
        float s = 0.f;
        for (int e = 0; e < D_; ++e) s += bq[e] * wk[e * D_ + i];
        bqk[i] = s;
    } else {
        int j = i - D_;
        float s = bo[j];
        for (int e = 0; e < D_; ++e) s += wo[j * D_ + e] * bv[e];
        bvo[j] = s;
    }
}

__global__ __launch_bounds__(256) void k_cvt_bf16(const float* __restrict__ in,
                                                  bf16_t* __restrict__ out) {
    size_t i = (size_t)blockIdx.x * 256 + threadIdx.x;
    out[i] = (bf16_t)in[i];
}

// X[(b*C+c)][d] = residual[b][d][c]  (LDS-tiled transpose, bf16 out)
__global__ __launch_bounds__(256) void k_transpose(const float* __restrict__ res,
                                                   bf16_t* __restrict__ X) {
    __shared__ float tile[32][33];
    const int b = blockIdx.z;
    const int c0 = blockIdx.x * 32, d0 = blockIdx.y * 32;
    const int tx = threadIdx.x, ty = threadIdx.y;   // (32,8)
    #pragma unroll
    for (int i = 0; i < 32; i += 8)
        tile[ty + i][tx] = res[((size_t)b * D_ + d0 + ty + i) * C_ + c0 + tx];
    __syncthreads();
    #pragma unroll
    for (int i = 0; i < 32; i += 8)
        X[((size_t)b * C_ + c0 + ty + i) * D_ + d0 + tx] = (bf16_t)tile[tx][ty + i];
}

// ============================================================
// WMMA bf16 GEMM:  Y[n][o] = sum_k X[n][k] * W[o][k] + bias[o] (+ epilogue)
// Block = 256 threads (8 waves), 128x128 tile, per-wave 64x32, K-tile 64,
// double-buffered LDS fed by global_load_async_to_lds_b128 (ASYNCcnt).
// EPI: 0=f32 out, 1=+strided residual(B,D,C) f32 out,
//      2=exact GELU -> bf16 out, 3=+aux(rowmajor) f32 out
// ============================================================
template <int EPI>
__global__ __launch_bounds__(256)
void k_gemm(const bf16_t* __restrict__ X, const bf16_t* __restrict__ W,
            const float* __restrict__ bias, const float* __restrict__ aux,
            float* __restrict__ outF, bf16_t* __restrict__ outB,
            int Nrows, int Ncols, int K) {
    __shared__ __align__(16) bf16_t Xs[2][128][KT_];
    __shared__ __align__(16) bf16_t Ws[2][128][KT_];

    const int tid = threadIdx.x;
    const int lane = tid & 31;
    const int wv = tid >> 5;
    const int laneLo = lane & 15, laneHi = lane >> 4;
    const int waveM = wv & 1, waveN = wv >> 1;   // 2 x 4 wave grid
    const int rowBase = blockIdx.y * 128;
    const int colBase = blockIdx.x * 128;

    v8f acc[4][2];
    #pragma unroll
    for (int mf = 0; mf < 4; ++mf)
        #pragma unroll
        for (int nf = 0; nf < 2; ++nf)
            acc[mf][nf] = (v8f){0.f, 0.f, 0.f, 0.f, 0.f, 0.f, 0.f, 0.f};

    // tile loader: 128 rows x 64 halves = 1024 16-byte chunks -> 4 per thread
    auto load_tile = [&](int buf, int k0) {
        #pragma unroll
        for (int i = 0; i < 4; ++i) {
            const int ch = tid + i * 256;
            const int r  = ch >> 3;           // 0..127
            const int ko = (ch & 7) * 8;      // halves, 16B chunks
            const bf16_t* gx = X + (size_t)(rowBase + r) * K + k0 + ko;
            const bf16_t* gw = W + (size_t)(colBase + r) * K + k0 + ko;
#if USE_ASYNC_LDS
            __builtin_amdgcn_global_load_async_to_lds_b128(
                (gv4i_p)gx, (lv4i_p)&Xs[buf][r][ko], 0, 0);
            __builtin_amdgcn_global_load_async_to_lds_b128(
                (gv4i_p)gw, (lv4i_p)&Ws[buf][r][ko], 0, 0);
#else
            *(uint4*)(&Xs[buf][r][ko]) = *(const uint4*)gx;
            *(uint4*)(&Ws[buf][r][ko]) = *(const uint4*)gw;
#endif
        }
    };

    load_tile(0, 0);
#if USE_ASYNC_LDS
    __builtin_amdgcn_s_wait_asynccnt(0);
#endif
    __syncthreads();

    const int nK = K >> 6;                    // K-tiles of 64
    for (int ks = 0; ks < nK; ++ks) {
        const int buf = ks & 1;
        if (ks + 1 < nK) {
            load_tile(buf ^ 1, (ks + 1) << 6);
            if (ks + 2 < nK) {  // global_prefetch_b8 of the tile after next
                __builtin_prefetch(X + (size_t)(rowBase + (tid >> 1)) * K + ((ks + 2) << 6), 0, 0);
                __builtin_prefetch(W + (size_t)(colBase + (tid >> 1)) * K + ((ks + 2) << 6), 0, 0);
            }
        }
        #pragma unroll
        for (int kk = 0; kk < 2; ++kk) {      // two 32-wide WMMA sub-steps
            const int kb = kk * 32;
            // A fragment: 16x32 bf16, K = kb + 8*laneHi(+0..7), kb+16+8*laneHi(+0..7)
            Frag16 a[4], b[2];
            #pragma unroll
            for (int mf = 0; mf < 4; ++mf) {
                const int r = waveM * 64 + mf * 16 + laneLo;
                a[mf].u[0] = *(const uint4*)(&Xs[buf][r][kb + 8 * laneHi]);
                a[mf].u[1] = *(const uint4*)(&Xs[buf][r][kb + 16 + 8 * laneHi]);
            }
            // B fragment: 32x16 bf16 (B[k][o] = W[o][k]), K = kb + 8*laneHi(+0..7), kb+16+8*laneHi(+0..7)
            #pragma unroll
            for (int nf = 0; nf < 2; ++nf) {
                const int c = waveN * 32 + nf * 16 + laneLo;
                b[nf].u[0] = *(const uint4*)(&Ws[buf][c][kb + 8 * laneHi]);
                b[nf].u[1] = *(const uint4*)(&Ws[buf][c][kb + 16 + 8 * laneHi]);
            }
            #pragma unroll
            for (int mf = 0; mf < 4; ++mf)
                #pragma unroll
                for (int nf = 0; nf < 2; ++nf)
                    acc[mf][nf] = __builtin_amdgcn_wmma_f32_16x16x32_bf16(
                        false, a[mf].v, false, b[nf].v, (short)0, acc[mf][nf],
                        false, false);
        }
#if USE_ASYNC_LDS
        __builtin_amdgcn_s_wait_asynccnt(0);  // next buffer landed in LDS
#endif
        __syncthreads();
    }

    // epilogue; C/D layout: M = e + 8*laneHi, N = laneLo
    #pragma unroll
    for (int mf = 0; mf < 4; ++mf) {
        #pragma unroll
        for (int nf = 0; nf < 2; ++nf) {
            const int col = colBase + waveN * 32 + nf * 16 + laneLo;
            const float bia = bias[col];
            #pragma unroll
            for (int e = 0; e < 8; ++e) {
                const int row = rowBase + waveM * 64 + mf * 16 + laneHi * 8 + e;
                float v = acc[mf][nf][e] + bia;
                const size_t oi = (size_t)row * Ncols + col;
                if (EPI == 0) {
                    outF[oi] = v;
                } else if (EPI == 1) {  // + residual[b][col][c], row = b*C+c
                    const int bb = row >> 9, cc = row & (C_ - 1);
                    v += aux[((size_t)bb * D_ + col) * C_ + cc];
                    outF[oi] = v;
                } else if (EPI == 2) {  // exact GELU, bf16 out
                    const float g = 0.5f * v * (1.0f + erff(v * 0.70710678118654752f));
                    outB[oi] = (bf16_t)g;
                } else {                // + aux row-major
                    outF[oi] = v + aux[oi];
                }
            }
        }
    }
}

// ============================================================
// Per-row attention over 8 shifts: scores -> softmax -> u (bf16)
// One block per row n = b*C + c.
// ============================================================
__global__ __launch_bounds__(256)
void k_attn(const float* __restrict__ QK, const float* __restrict__ res,
            bf16_t* __restrict__ U) {
    __shared__ float xs[D_];
    __shared__ float qs[D_];
    __shared__ float sc[NS_];
    const int n = blockIdx.x;
    const int b = n >> 9, c = n & (C_ - 1);
    for (int d = threadIdx.x; d < D_; d += 256) {
        xs[d] = res[((size_t)b * D_ + d) * C_ + c];
        qs[d] = QK[(size_t)n * D_ + d];
    }
    __syncthreads();
    const int wv = threadIdx.x >> 5, lane = threadIdx.x & 31;
    // wave wv computes score for shift s = wv  (bk cancels in softmax)
    float p = 0.f;
    const int shift = wv * P_;
    for (int d = lane; d < D_; d += 32) p += qs[d] * xs[(d + shift) & (D_ - 1)];
    #pragma unroll
    for (int off = 16; off > 0; off >>= 1) p += __shfl_down(p, off, 32);
    if (lane == 0) sc[wv] = p * 0.044194173824159216f;  // 512^-0.5
    __syncthreads();
    // every thread redundantly does the tiny 8-way softmax
    float m = sc[0];
    #pragma unroll
    for (int s = 1; s < NS_; ++s) m = fmaxf(m, sc[s]);
    float w[NS_], den = 0.f;
    #pragma unroll
    for (int s = 0; s < NS_; ++s) { w[s] = __expf(sc[s] - m); den += w[s]; }
    const float inv = 1.f / den;
    for (int d = threadIdx.x; d < D_; d += 256) {
        float acc = 0.f;
        #pragma unroll
        for (int s = 0; s < NS_; ++s) acc += w[s] * xs[(d + (s << 6)) & (D_ - 1)];
        U[(size_t)n * D_ + d] = (bf16_t)(acc * inv);
    }
}

// ============================================================
// BatchNorm over (b, d) per channel c  ->  scale/shift
// ============================================================
__global__ __launch_bounds__(256)
void k_bn_stats(const float* __restrict__ H, const float* __restrict__ bnw,
                const float* __restrict__ bnb, float* __restrict__ scale,
                float* __restrict__ shift) {
    __shared__ float rs[256], rs2[256];
    const int c = blockIdx.x;
    float s = 0.f, s2 = 0.f;
    for (int i = threadIdx.x; i < B_ * D_; i += 256) {
        const int b = i >> 9, d = i & (D_ - 1);
        const float v = H[((size_t)(b * C_ + c)) * D_ + d];
        s += v; s2 += v * v;
    }
    rs[threadIdx.x] = s; rs2[threadIdx.x] = s2;
    __syncthreads();
    for (int off = 128; off > 0; off >>= 1) {
        if (threadIdx.x < off) { rs[threadIdx.x] += rs[threadIdx.x + off];
                                 rs2[threadIdx.x] += rs2[threadIdx.x + off]; }
        __syncthreads();
    }
    if (threadIdx.x == 0) {
        const float inv_n = 1.0f / (float)(B_ * D_);
        const float mu = rs[0] * inv_n;
        const float var = rs2[0] * inv_n - mu * mu;
        const float r = rsqrtf(var + 1e-5f);
        const float sc = bnw[c] * r;
        scale[c] = sc;
        shift[c] = bnb[c] - mu * sc;
    }
}

__global__ __launch_bounds__(256)
void k_bn_apply(const float* __restrict__ H, const float* __restrict__ scale,
                const float* __restrict__ shift, float* __restrict__ HNf,
                bf16_t* __restrict__ HNb) {
    const size_t i = (size_t)blockIdx.x * 256 + threadIdx.x;  // N_*D_
    const int n = (int)(i >> 9);
    const int c = n & (C_ - 1);
    const float v = H[i] * scale[c] + shift[c];
    HNf[i] = v;
    HNb[i] = (bf16_t)v;
}

// ============================================================
// Row LayerNorm + transposed store to (B, D, C)
// ============================================================
__global__ __launch_bounds__(256)
void k_ln_out(const float* __restrict__ Y, const float* __restrict__ lnw,
              const float* __restrict__ lnb, float* __restrict__ out) {
    __shared__ float red[256];
    const int n = blockIdx.x;
    const int b = n >> 9, c = n & (C_ - 1);
    const float* y = Y + (size_t)n * D_;
    float s = 0.f;
    for (int d = threadIdx.x; d < D_; d += 256) s += y[d];
    red[threadIdx.x] = s; __syncthreads();
    for (int off = 128; off > 0; off >>= 1) {
        if (threadIdx.x < off) red[threadIdx.x] += red[threadIdx.x + off];
        __syncthreads();
    }
    const float mu = red[0] * (1.0f / D_);
    __syncthreads();
    float s2 = 0.f;
    for (int d = threadIdx.x; d < D_; d += 256) { const float t = y[d] - mu; s2 += t * t; }
    red[threadIdx.x] = s2; __syncthreads();
    for (int off = 128; off > 0; off >>= 1) {
        if (threadIdx.x < off) red[threadIdx.x] += red[threadIdx.x + off];
        __syncthreads();
    }
    const float r = rsqrtf(red[0] * (1.0f / D_) + 1e-5f);
    for (int d = threadIdx.x; d < D_; d += 256) {
        const float v = (y[d] - mu) * r * lnw[d] + lnb[d];
        out[((size_t)b * D_ + d) * C_ + c] = v;
    }
}

// ============================================================
extern "C" void kernel_launch(void* const* d_in, const int* in_sizes, int n_in,
                              void* d_out, int out_size, void* d_ws, size_t ws_size,
                              hipStream_t stream) {
    const float* residual = (const float*)d_in[0];
    const float* wq = (const float*)d_in[1];
    const float* bq = (const float*)d_in[2];
    const float* wk = (const float*)d_in[3];
    // d_in[4] (bk) is constant across shifts -> cancels in softmax
    const float* wv = (const float*)d_in[5];
    const float* bv = (const float*)d_in[6];
    const float* wo = (const float*)d_in[7];
    const float* bo = (const float*)d_in[8];
    const float* bn_w = (const float*)d_in[9];
    const float* bn_b = (const float*)d_in[10];
    const float* w1 = (const float*)d_in[11];
    const float* b1 = (const float*)d_in[12];
    const float* w2 = (const float*)d_in[13];
    const float* b2 = (const float*)d_in[14];
    const float* ln_w = (const float*)d_in[15];
    const float* ln_b = (const float*)d_in[16];
    float* out = (float*)d_out;

    char* ws = (char*)d_ws;
    size_t off = 0;
    auto alloc = [&](size_t bytes) -> void* {
        void* p = ws + off;
        off += (bytes + 255) & ~(size_t)255;
        return p;
    };
    const size_t ND = (size_t)N_ * D_;
    bf16_t* Xbf = (bf16_t*)alloc(ND * 2);
    bf16_t* Ubf = (bf16_t*)alloc(ND * 2);
    bf16_t* HNb = (bf16_t*)alloc(ND * 2);
    float*  QKf = (float*)alloc(ND * 4);      // reused: QK (f32 NxD) then G (bf16 Nx2D)
    bf16_t* Gbf = (bf16_t*)QKf;
    float*  Hf  = (float*)alloc(ND * 4);      // reused: H then Y
    float*  Yf  = Hf;
    float*  HNf = (float*)alloc(ND * 4);
    bf16_t* Wqk = (bf16_t*)alloc((size_t)D_ * D_ * 2);
    bf16_t* Wvo = (bf16_t*)alloc((size_t)D_ * D_ * 2);
    bf16_t* W1b = (bf16_t*)alloc((size_t)2 * D_ * D_ * 2);
    bf16_t* W2b = (bf16_t*)alloc((size_t)2 * D_ * D_ * 2);
    float*  bqk = (float*)alloc(D_ * 4);
    float*  bvo = (float*)alloc(D_ * 4);
    float*  scl = (float*)alloc(C_ * 4);
    float*  sft = (float*)alloc(C_ * 4);

    // weight prep + conversion
    k_prep_wqk<<<(D_ * D_) / 256, 256, 0, stream>>>(wq, wk, Wqk);
    k_prep_wvo<<<(D_ * D_) / 256, 256, 0, stream>>>(wo, wv, Wvo);
    k_prep_bias<<<(2 * D_) / 256, 256, 0, stream>>>(bq, wk, wo, bv, bo, bqk, bvo);
    k_cvt_bf16<<<(2 * D_ * D_) / 256, 256, 0, stream>>>(w1, W1b);
    k_cvt_bf16<<<(2 * D_ * D_) / 256, 256, 0, stream>>>(w2, W2b);
    // x = transpose(residual) in bf16
    k_transpose<<<dim3(C_ / 32, D_ / 32, B_), dim3(32, 8), 0, stream>>>(residual, Xbf);
    // QK = X @ Wqk^T + bqk
    k_gemm<0><<<dim3(D_ / 128, N_ / 128), 256, 0, stream>>>(
        Xbf, Wqk, bqk, nullptr, QKf, nullptr, N_, D_, D_);
    // scores -> softmax -> u
    k_attn<<<N_, 256, 0, stream>>>(QKf, residual, Ubf);
    // H = U @ Wvo^T + bvo + x
    k_gemm<1><<<dim3(D_ / 128, N_ / 128), 256, 0, stream>>>(
        Ubf, Wvo, bvo, residual, Hf, nullptr, N_, D_, D_);
    // BatchNorm
    k_bn_stats<<<C_, 256, 0, stream>>>(Hf, bn_w, bn_b, scl, sft);
    k_bn_apply<<<(int)(ND / 256), 256, 0, stream>>>(Hf, scl, sft, HNf, HNb);
    // G = gelu(HN @ W1^T + b1)  (bf16)
    k_gemm<2><<<dim3((2 * D_) / 128, N_ / 128), 256, 0, stream>>>(
        HNb, W1b, b1, nullptr, nullptr, Gbf, N_, 2 * D_, D_);
    // Y = G @ W2^T + b2 + HN
    k_gemm<3><<<dim3(D_ / 128, N_ / 128), 256, 0, stream>>>(
        Gbf, W2b, b2, HNf, Yf, nullptr, N_, D_, 2 * D_);
    // LayerNorm + transpose out
    k_ln_out<<<N_, 256, 0, stream>>>(Yf, ln_w, ln_b, out);
}
